// Attention_74174085202554
// MI455X (gfx1250) — matrix operand order repaired and checked by reading, not compiled, not run on hardware
//
#include <hip/hip_runtime.h>

typedef __bf16 bf16;
typedef __attribute__((ext_vector_type(16))) __bf16 v16bf;
typedef __attribute__((ext_vector_type(8)))  float  v8f;

#define SEQ   1024
#define DIM   768
#define HEADS 12
#define HDIM  64
#define BATCH 8
#define QKSCALE 0.125f   // 64^-0.5

// ---------------------------------------------------------------- WMMA helper
__device__ __forceinline__ v8f wmma_bf16(v16bf a, v16bf b, v8f c) {
  // (neg_a, A, neg_b, B, c_mod, C, reuse_a, reuse_b)
  return __builtin_amdgcn_wmma_f32_16x16x32_bf16(false, a, false, b, (short)0, c,
                                                 false, false);
}

__device__ __forceinline__ v8f v8f_zero() {
  v8f z = {0.f, 0.f, 0.f, 0.f, 0.f, 0.f, 0.f, 0.f};
  return z;
}

// ------------------------------------------------- CDNA5 async global -> LDS
// One GLOBAL_LOAD_ASYNC_TO_LDS_B128 per lane (GVS addressing: sgpr base +
// per-lane byte offset). Tracked by ASYNCcnt, waited with s_wait_asynccnt.
__device__ __forceinline__ void async_copy_b128(uint32_t lds_byte,
                                                uint32_t gbyte_off,
                                                uint64_t gbase) {
  asm volatile("global_load_async_to_lds_b128 %0, %1, %2"
               :: "v"(lds_byte), "v"(gbyte_off), "s"(gbase)
               : "memory");
}
__device__ __forceinline__ void wait_async0() {
  asm volatile("s_wait_asynccnt 0x0" ::: "memory");
}
__device__ __forceinline__ void wait_async1() {
  asm volatile("s_wait_asynccnt 0x1" ::: "memory");
}
__device__ __forceinline__ void wait_async2() {
  asm volatile("s_wait_asynccnt 0x2" ::: "memory");
}

// A-fragment 16x32 bf16, row-major src (lda elements).
// lane l -> row M=l%16, K-half kh=l/16; two contiguous 16B chunks at +0, +16.
__device__ __forceinline__ v16bf frag_a16(const bf16* base, int lda, int lane) {
  const int m  = lane & 15;
  const int kh = (lane >> 4) & 1;
  const bf16* p = base + (size_t)m * lda + kh * 8;
  union { v16bf v; uint4 q[2]; } u;
  u.q[0] = *(const uint4*)(p);
  u.q[1] = *(const uint4*)(p + 16);
  return u.v;
}

// A-fragment with on-the-fly f32 -> bf16 conversion (fp32 activations)
__device__ __forceinline__ v16bf frag_a_f32(const float* base, int lda, int lane) {
  const int m  = lane & 15;
  const int kh = (lane >> 4) & 1;
  const float* p = base + (size_t)m * lda + kh * 8;
  union { float4 v; float f[4]; } c0, c1, c2, c3;
  c0.v = *(const float4*)(p);
  c1.v = *(const float4*)(p + 4);
  c2.v = *(const float4*)(p + 16);
  c3.v = *(const float4*)(p + 20);
  v16bf a;
#pragma unroll
  for (int e = 0; e < 4; ++e) {
    a[e]      = (bf16)c0.f[e];
    a[4 + e]  = (bf16)c1.f[e];
    a[8 + e]  = (bf16)c2.f[e];
    a[12 + e] = (bf16)c3.f[e];
  }
  return a;
}

// B-fragment 32x16 bf16 from "B^T" row-major src [N rows x K cols]:
// lane l -> col N=l%16, K-half kh=l/16; two contiguous 16B chunks at +0, +16.
__device__ __forceinline__ v16bf frag_b16(const bf16* base, int ldb, int lane) {
  const int n  = lane & 15;
  const int kh = (lane >> 4) & 1;
  const bf16* p = base + (size_t)n * ldb + kh * 8;
  union { v16bf v; uint4 q[2]; } u;
  u.q[0] = *(const uint4*)(p);
  u.q[1] = *(const uint4*)(p + 16);
  return u.v;
}

// ---------------------------------------------------------------- fp32->bf16
__global__ __launch_bounds__(256) void cvt_kernel(const float* __restrict__ src,
                                                  bf16* __restrict__ dst, int n) {
  int i = (blockIdx.x * 256 + threadIdx.x) * 4;
  if (i + 3 < n) {
    float4 f = *(const float4*)(src + i);
    dst[i + 0] = (bf16)f.x;
    dst[i + 1] = (bf16)f.y;
    dst[i + 2] = (bf16)f.z;
    dst[i + 3] = (bf16)f.w;
  }
}

// ---------------------------------------------------------------- fused QKV
// X:[8192 x 768] f32 row-major, W:[2304 x 768] bf16 (row-major = B^T).
// Weight tiles (64x32) are async-DMAd to LDS, double buffered, shared by the
// 8 waves of the block. Epilogue scatters q (scaled) / k / v^T.
__global__ __launch_bounds__(256) void qkv_kernel(const float* __restrict__ X,
                                                  const bf16* __restrict__ W,
                                                  bf16* __restrict__ qb,
                                                  bf16* __restrict__ kb,
                                                  bf16* __restrict__ vT) {
  __shared__ bf16 Bt[2][64][32];   // 2 x 4KB weight tiles

  const int t    = threadIdx.x;
  const int lane = t & 31;
  const int wave = t >> 5;
  const int m0 = blockIdx.x * 128 + wave * 16;  // M tile (of 8192)
  const int n0 = blockIdx.y * 64;               // N tile (of 2304)

  // cooperative B staging: 64 rows x 4 chunks of 16B -> 256 lanes, 1 b128 each
  const int crow = t >> 2;
  const int cch  = t & 3;
  const uint32_t ldsB  = (uint32_t)(size_t)(&Bt[0][0][0]);
  const uint32_t bdst  = (uint32_t)(crow * 64 + cch * 16);           // bytes
  const uint32_t BSTRIDE = 64u * 32u * 2u;                           // 4096
  const uint64_t wbase = (uint64_t)W;
  const uint32_t gsrc  = ((uint32_t)(n0 + crow) * DIM + cch * 8) * 2; // + k0*2

  async_copy_b128(ldsB + bdst, gsrc, wbase);   // prologue: tile 0 -> buf 0

  v8f acc[4];
#pragma unroll
  for (int c = 0; c < 4; ++c) acc[c] = v8f_zero();

  const float* xrow = X + (size_t)m0 * DIM;
  const int NK = DIM / 32;   // 24
  for (int i = 0; i < NK; ++i) {
    const int k0 = i * 32;
    if (i + 1 < NK) {
      async_copy_b128(ldsB + (uint32_t)((i + 1) & 1) * BSTRIDE + bdst,
                      gsrc + (uint32_t)(k0 + 32) * 2, wbase);
      wait_async1();          // tile i has landed (in-order completion)
    } else {
      wait_async0();
    }
    __syncthreads();

    v16bf a = frag_a_f32(xrow + k0, DIM, lane);
    const bf16* Bcur = &Bt[i & 1][0][0];
#pragma unroll
    for (int c = 0; c < 4; ++c) {
      acc[c] = wmma_bf16(a, frag_b16(Bcur + c * 16 * 32, 32, lane), acc[c]);
    }
    __syncthreads();          // everyone done reading before next overwrite
  }

  const int g  = lane >> 4;
  const int nc = lane & 15;
#pragma unroll
  for (int c = 0; c < 4; ++c) {
#pragma unroll
    for (int r = 0; r < 8; ++r) {
      float v = acc[c][r];
      const int m = m0 + r + 8 * g;          // global row in [0,8192)
      const int n = n0 + c * 16 + nc;        // global col in [0,2304)
      const int part = n / DIM;              // 0=q 1=k 2=v
      const int rem  = n - part * DIM;
      const int h = rem >> 6;
      const int d = rem & 63;
      const int b = m >> 10;
      const int s = m & (SEQ - 1);
      const int bh = b * HEADS + h;
      if (part == 0) {
        qb[((size_t)bh * SEQ + s) * HDIM + d] = (bf16)(v * QKSCALE);
      } else if (part == 1) {
        kb[((size_t)bh * SEQ + s) * HDIM + d] = (bf16)v;
      } else {
        vT[((size_t)bh * HDIM + d) * SEQ + s] = (bf16)v;
      }
    }
  }
}

// ---------------------------------------------------------------- attention
// One wave = 16 query rows; the block's 8 waves share async-staged K (32x64)
// and V^T (64x32) tiles, double buffered. Online softmax; P re-laid out via
// per-wave LDS. q,k:[96][1024][64] (q pre-scaled), vT:[96][64][1024].
__global__ __launch_bounds__(256) void attn_kernel(const bf16* __restrict__ q,
                                                   const bf16* __restrict__ k,
                                                   const bf16* __restrict__ vT,
                                                   bf16* __restrict__ aout) {
  __shared__ bf16 Kt[2][32][64];    // 2 x 4KB
  __shared__ bf16 Vt[2][64][32];    // 2 x 4KB
  __shared__ bf16 pbuf[8][16][32];  // per-wave P staging, 8KB

  const int t    = threadIdx.x;
  const int lane = t & 31;
  const int wave = t >> 5;
  const int bh = blockIdx.x;        // 0..95
  const int b  = bh / HEADS;
  const int h  = bh - b * HEADS;
  const int q0 = blockIdx.y * 128 + wave * 16;

  // K staging: 32 rows x 8 chunks of 16B
  const int krow = t >> 3, kch = t & 7;
  const uint32_t ldsK = (uint32_t)(size_t)(&Kt[0][0][0]);
  const uint32_t kdst = (uint32_t)(krow * 128 + kch * 16);
  const uint64_t kbase = (uint64_t)k;
  const uint32_t kg = (((uint32_t)bh * SEQ + krow) * HDIM + kch * 8) * 2;
  const uint32_t KBUF = 32u * 64u * 2u;  // 4096
  // V staging: 64 rows x 4 chunks of 16B
  const int vrow = t >> 2, vch = t & 3;
  const uint32_t ldsV = (uint32_t)(size_t)(&Vt[0][0][0]);
  const uint32_t vdst = (uint32_t)(vrow * 64 + vch * 16);
  const uint64_t vbase = (uint64_t)vT;
  const uint32_t vg = (((uint32_t)bh * HDIM + vrow) * SEQ + vch * 8) * 2;
  const uint32_t VBUF = 64u * 32u * 2u;  // 4096

  async_copy_b128(ldsK + kdst, kg, kbase);   // tile 0
  async_copy_b128(ldsV + vdst, vg, vbase);

  const bf16* qrow = q + ((size_t)bh * SEQ + q0) * HDIM;
  const v16bf qa0 = frag_a16(qrow, HDIM, lane);        // head-dim K = 0..31
  const v16bf qa1 = frag_a16(qrow + 32, HDIM, lane);   // head-dim K = 32..63

  v8f O[4];
#pragma unroll
  for (int c = 0; c < 4; ++c) O[c] = v8f_zero();
  float mrun[8], lrun[8];
#pragma unroll
  for (int r = 0; r < 8; ++r) { mrun[r] = -1e30f; lrun[r] = 0.f; }

  const int g  = lane >> 4;
  const int nc = lane & 15;

  const int NJ = SEQ / 32;   // 32 key steps
  for (int i = 0; i < NJ; ++i) {
    const int j0 = i * 32;
    if (i + 1 < NJ) {
      const uint32_t nb = (uint32_t)((i + 1) & 1);
      async_copy_b128(ldsK + nb * KBUF + kdst,
                      kg + (uint32_t)(j0 + 32) * HDIM * 2, kbase);
      async_copy_b128(ldsV + nb * VBUF + vdst,
                      vg + (uint32_t)(j0 + 32) * 2, vbase);
      wait_async2();          // current tile's two DMAs have landed
    } else {
      wait_async0();
    }
    __syncthreads();

    const bf16* Kc = &Kt[i & 1][0][0];
    const bf16* Vc = &Vt[i & 1][0][0];

    // scores for key columns [j0, j0+32)
    v8f s0 = wmma_bf16(qa0, frag_b16(Kc, HDIM, lane), v8f_zero());
    s0     = wmma_bf16(qa1, frag_b16(Kc + 32, HDIM, lane), s0);
    v8f s1 = wmma_bf16(qa0, frag_b16(Kc + 16 * HDIM, HDIM, lane), v8f_zero());
    s1     = wmma_bf16(qa1, frag_b16(Kc + 16 * HDIM + 32, HDIM, lane), s1);

    // online softmax (row r+8g spread across 16 lanes of this half)
    float p0[8], p1[8], alpha[8];
#pragma unroll
    for (int r = 0; r < 8; ++r) {
      float cm = fmaxf(s0[r], s1[r]);
#pragma unroll
      for (int off = 1; off < 16; off <<= 1)
        cm = fmaxf(cm, __shfl_xor(cm, off, 32));
      const float nm = fmaxf(mrun[r], cm);
      const float al = __expf(mrun[r] - nm);
      mrun[r] = nm;
      const float e0 = __expf(s0[r] - nm);
      const float e1 = __expf(s1[r] - nm);
      p0[r] = e0; p1[r] = e1;
      float rs = e0 + e1;
#pragma unroll
      for (int off = 1; off < 16; off <<= 1)
        rs += __shfl_xor(rs, off, 32);
      lrun[r] = lrun[r] * al + rs;
      alpha[r] = al;
    }
#pragma unroll
    for (int c = 0; c < 4; ++c)
#pragma unroll
      for (int r = 0; r < 8; ++r) O[c][r] *= alpha[r];

    // stage P (16x32) to per-wave LDS, reload in A-fragment layout
#pragma unroll
    for (int r = 0; r < 8; ++r) {
      pbuf[wave][r + 8 * g][nc]      = (bf16)p0[r];
      pbuf[wave][r + 8 * g][16 + nc] = (bf16)p1[r];
    }
    const v16bf pf = frag_a16(&pbuf[wave][0][0], 32, lane);

    // O += P * V
#pragma unroll
    for (int c = 0; c < 4; ++c) {
      O[c] = wmma_bf16(pf, frag_b16(Vc + c * 16 * 32, 32, lane), O[c]);
    }
    __syncthreads();          // all waves done with K/V bufs before overwrite
  }

  // normalize + write [B,N,C] bf16
#pragma unroll
  for (int r = 0; r < 8; ++r) {
    const float inv = 1.0f / lrun[r];
    const int m = r + 8 * g;
#pragma unroll
    for (int c = 0; c < 4; ++c) {
      aout[((size_t)(b * SEQ) + q0 + m) * DIM + h * HDIM + c * 16 + nc] =
          (bf16)(O[c][r] * inv);
    }
  }
}

// ---------------------------------------------------------------- projection
// A:[8192 x 768] bf16, W:[768 x 768] bf16 (B^T form), bias f32, out f32.
__global__ __launch_bounds__(256) void proj_kernel(const bf16* __restrict__ A,
                                                   const bf16* __restrict__ W,
                                                   const float* __restrict__ bias,
                                                   float* __restrict__ out) {
  __shared__ bf16 Bt[2][64][32];

  const int t    = threadIdx.x;
  const int lane = t & 31;
  const int wave = t >> 5;
  const int m0 = blockIdx.x * 128 + wave * 16;
  const int n0 = blockIdx.y * 64;

  const int crow = t >> 2;
  const int cch  = t & 3;
  const uint32_t ldsB = (uint32_t)(size_t)(&Bt[0][0][0]);
  const uint32_t bdst = (uint32_t)(crow * 64 + cch * 16);
  const uint32_t BSTRIDE = 64u * 32u * 2u;
  const uint64_t wbase = (uint64_t)W;
  const uint32_t gsrc = ((uint32_t)(n0 + crow) * DIM + cch * 8) * 2;

  async_copy_b128(ldsB + bdst, gsrc, wbase);

  v8f acc[4];
#pragma unroll
  for (int c = 0; c < 4; ++c) acc[c] = v8f_zero();

  const bf16* arow = A + (size_t)m0 * DIM;
  const int NK = DIM / 32;
  for (int i = 0; i < NK; ++i) {
    const int k0 = i * 32;
    if (i + 1 < NK) {
      async_copy_b128(ldsB + (uint32_t)((i + 1) & 1) * BSTRIDE + bdst,
                      gsrc + (uint32_t)(k0 + 32) * 2, wbase);
      wait_async1();
    } else {
      wait_async0();
    }
    __syncthreads();

    v16bf a = frag_a16(arow + k0, DIM, lane);
    const bf16* Bcur = &Bt[i & 1][0][0];
#pragma unroll
    for (int c = 0; c < 4; ++c) {
      acc[c] = wmma_bf16(a, frag_b16(Bcur + c * 16 * 32, 32, lane), acc[c]);
    }
    __syncthreads();
  }

  const int g  = lane >> 4;
  const int nc = lane & 15;
#pragma unroll
  for (int c = 0; c < 4; ++c) {
    const int n = n0 + c * 16 + nc;
    const float bv = bias[n];
#pragma unroll
    for (int r = 0; r < 8; ++r) {
      const int m = m0 + r + 8 * g;
      out[(size_t)m * DIM + n] = acc[c][r] + bv;
    }
  }
}

// ---------------------------------------------------------------- launcher
extern "C" void kernel_launch(void* const* d_in, const int* in_sizes, int n_in,
                              void* d_out, int out_size, void* d_ws, size_t ws_size,
                              hipStream_t stream) {
  const float* x        = (const float*)d_in[0];
  const float* y        = (const float*)d_in[1];
  const float* x_qkv_w  = (const float*)d_in[2];
  const float* y_qkv_w  = (const float*)d_in[3];
  const float* x_proj_w = (const float*)d_in[4];
  const float* x_proj_b = (const float*)d_in[5];
  const float* y_proj_w = (const float*)d_in[6];
  const float* y_proj_b = (const float*)d_in[7];
  float* out = (float*)d_out;

  char* ws = (char*)d_ws;
  size_t off = 0;
  auto alloc_bf16 = [&](size_t elems) -> bf16* {
    bf16* p = (bf16*)(ws + off);
    off += ((elems * sizeof(bf16) + 255) / 256) * 256;
    return p;
  };

  const size_t QKV_W = (size_t)3 * DIM * DIM;                 // 2304*768
  const size_t PRJ_W = (size_t)DIM * DIM;                     // 768*768
  const size_t HEADBUF = (size_t)BATCH * HEADS * SEQ * HDIM;  // 96*1024*64
  const size_t ACT = (size_t)BATCH * SEQ * DIM;               // 8*1024*768

  bf16* wqx = alloc_bf16(QKV_W);
  bf16* wqy = alloc_bf16(QKV_W);
  bf16* wpx = alloc_bf16(PRJ_W);
  bf16* wpy = alloc_bf16(PRJ_W);
  bf16* qx  = alloc_bf16(HEADBUF);
  bf16* kx  = alloc_bf16(HEADBUF);
  bf16* vTx = alloc_bf16(HEADBUF);
  bf16* qy  = alloc_bf16(HEADBUF);
  bf16* ky  = alloc_bf16(HEADBUF);
  bf16* vTy = alloc_bf16(HEADBUF);
  bf16* aox = alloc_bf16(ACT);
  bf16* aoy = alloc_bf16(ACT);

  // 1) weights -> bf16 (small; lives in L2)
  cvt_kernel<<<(int)(QKV_W / 1024), 256, 0, stream>>>(x_qkv_w, wqx, (int)QKV_W);
  cvt_kernel<<<(int)(QKV_W / 1024), 256, 0, stream>>>(y_qkv_w, wqy, (int)QKV_W);
  cvt_kernel<<<(int)(PRJ_W / 1024), 256, 0, stream>>>(x_proj_w, wpx, (int)PRJ_W);
  cvt_kernel<<<(int)(PRJ_W / 1024), 256, 0, stream>>>(y_proj_w, wpy, (int)PRJ_W);

  // 2) fused QKV GEMM + head scatter
  dim3 gq(8192 / 128, (3 * DIM) / 64);
  qkv_kernel<<<gq, 256, 0, stream>>>(x, wqx, qx, kx, vTx);
  qkv_kernel<<<gq, 256, 0, stream>>>(y, wqy, qy, ky, vTy);

  // 3) flash attention with swapped V: x-branch uses y_v, y-branch uses x_v
  dim3 ga(BATCH * HEADS, SEQ / 128);
  attn_kernel<<<ga, 256, 0, stream>>>(qx, kx, vTy, aox);
  attn_kernel<<<ga, 256, 0, stream>>>(qy, ky, vTx, aoy);

  // 4) output projections (f32 out, bias add)
  dim3 gp(8192 / 128, DIM / 64);
  proj_kernel<<<gp, 256, 0, stream>>>(aox, wpx, x_proj_b, out);
  proj_kernel<<<gp, 256, 0, stream>>>(aoy, wpy, y_proj_b, out + ACT);
}
